// BayesianLinear_36223754174952
// MI455X (gfx1250) — matrix operand order repaired and checked by reading, not compiled, not run on hardware
//
#include <hip/hip_runtime.h>
#include <hip/hip_bf16.h>

// ---------------------------------------------------------------------------
// BayesianLinear forward for MI455X (gfx1250, wave32, WMMA).
//   W    = weight_mu + softplus(weight_rho) * weight_eps     (bf16, in d_ws)
//   bias = bias_mu   + softplus(bias_rho)   * bias_eps       (f32,  in d_ws)
//   out  = x @ W^T + bias                                    (f32)
// GEMM: M=8192, N=4096, K=4096. bf16 WMMA (16x16x32) with f32 accumulation.
// B tile staged via GLOBAL_LOAD_ASYNC_TO_LDS (ASYNCcnt) when available.
// ---------------------------------------------------------------------------

typedef __bf16 bf16_t;
typedef __attribute__((ext_vector_type(16))) __bf16 v16bf;
typedef __attribute__((ext_vector_type(8)))  __bf16 v8bf;
typedef __attribute__((ext_vector_type(4)))  __bf16 v4bf;
typedef __attribute__((ext_vector_type(8)))  float  v8f;
typedef __attribute__((ext_vector_type(4)))  float  v4f;
typedef int v4i __attribute__((ext_vector_type(4)));

#define IN_F  4096
#define OUT_F 4096
#define BATCH 8192

#if __has_builtin(__builtin_amdgcn_global_load_async_to_lds_b128)
#define HAS_ASYNC_LDS 1
#else
#define HAS_ASYNC_LDS 0
#endif

// Builtin signature (from hipcc diagnostic): param1 = AS1 v4i*, param2 = AS3 v4i*
typedef __attribute__((address_space(1))) v4i* gas_v4i_p;
typedef __attribute__((address_space(3))) v4i* las_v4i_p;

__device__ __forceinline__ void wait_async_all() {
#if __has_builtin(__builtin_amdgcn_s_wait_asynccnt)
    __builtin_amdgcn_s_wait_asynccnt(0);
#else
    asm volatile("s_wait_asynccnt 0x0" ::: "memory");
#endif
}

// ---------------- weight / bias sampling ----------------

__device__ __forceinline__ float softplus_f(float r) {
    // softplus(r) = log1p(exp(r)); for large r it is r (avoids overflow)
    return (r > 15.0f) ? r : log1pf(__expf(r));
}

__global__ __launch_bounds__(256) void prep_weight_kernel(
    const float* __restrict__ mu, const float* __restrict__ rho,
    const float* __restrict__ eps, bf16_t* __restrict__ w)
{
    size_t i = ((size_t)blockIdx.x * 256 + threadIdx.x) * 4;
    v4f m = *(const v4f*)(mu  + i);
    v4f r = *(const v4f*)(rho + i);
    v4f e = *(const v4f*)(eps + i);
    v4f s;
#pragma unroll
    for (int j = 0; j < 4; ++j) {
        s[j] = fmaf(softplus_f(r[j]), e[j], m[j]);
    }
    *(v4bf*)(w + i) = __builtin_convertvector(s, v4bf);
}

__global__ __launch_bounds__(256) void prep_bias_kernel(
    const float* __restrict__ mu, const float* __restrict__ rho,
    const float* __restrict__ eps, float* __restrict__ b)
{
    int i = blockIdx.x * 256 + threadIdx.x;
    if (i < OUT_F) {
        b[i] = fmaf(softplus_f(rho[i]), eps[i], mu[i]);
    }
}

// ---------------- WMMA GEMM ----------------

__device__ __forceinline__ v16bf mk16(v8bf lo, v8bf hi) {
    union { v16bf v; v8bf h[2]; } u;
    u.h[0] = lo; u.h[1] = hi;
    return u.v;
}

// Block tile: 128(M) x 128(N), BK = 32. 256 threads = 8 wave32 waves.
// Wave grid 4(M) x 2(N); each wave owns 32x64 = 2x4 WMMA 16x16 tiles.
__global__ __launch_bounds__(256) void gemm_wmma_kernel(
    const float*  __restrict__ X,     // [BATCH, IN_F] f32, row-major
    const bf16_t* __restrict__ W,     // [OUT_F, IN_F] bf16, row-major (N-major)
    const float*  __restrict__ bias,  // [OUT_F] f32
    float*        __restrict__ out)   // [BATCH, OUT_F] f32
{
    __shared__ bf16_t lds_a[2][128][32];  // [buf][m][k]  8KB each
    __shared__ bf16_t lds_b[2][128][32];  // [buf][n][k]  8KB each

    const int tid  = threadIdx.x;
    const int lane = tid & 31;
    const int wave = tid >> 5;
    const int wm   = wave >> 1;   // 0..3 -> m offset wm*32
    const int wn   = wave & 1;    // 0..1 -> n offset wn*64
    const int lrow = lane & 15;
    const int lhi  = lane >> 4;   // 0/1 half-wave select

    const int m0 = blockIdx.y * 128;
    const int n0 = blockIdx.x * 128;

    // Cooperative A-tile loading: 2 threads per row, 16 f32 each.
    const int ldrow = tid >> 1;          // 0..127
    const int ldseg = (tid & 1) * 16;    // 0 or 16

    const float* xrow = X + (size_t)(m0 + ldrow) * IN_F + ldseg;

#if HAS_ASYNC_LDS
    // Async B-tile loading: 512 x 16B chunks; each thread issues 2 b128s.
    // chunk ch: row = ch>>2 (64B rows), seg = (ch&3)*8 bf16.
    const int b_row0 = tid >> 2;          // 0..63
    const int b_seg0 = (tid & 3) * 8;     // 0,8,16,24
    const bf16_t* wsrc0 = W + (size_t)(n0 + b_row0)      * IN_F + b_seg0;
    const bf16_t* wsrc1 = W + (size_t)(n0 + b_row0 + 64) * IN_F + b_seg0;
#else
    const bf16_t* wrow = W + (size_t)(n0 + ldrow) * IN_F + ldseg;
    uint4 b_reg[2];   // 16 bf16 of W (32 bytes)
#endif

    v8f acc[2][4];
    const v8f vzero = {0.f,0.f,0.f,0.f,0.f,0.f,0.f,0.f};
#pragma unroll
    for (int t = 0; t < 2; ++t)
#pragma unroll
        for (int u = 0; u < 4; ++u) acc[t][u] = vzero;

    v4f a_reg[4];   // 16 f32 of x

    // Issue loads for k-step starting at k0; B goes straight to lds_b[nbuf].
    auto load_stage = [&](int k0, int nbuf) {
#pragma unroll
        for (int i = 0; i < 4; ++i)
            a_reg[i] = *(const v4f*)(xrow + k0 + i * 4);
#if HAS_ASYNC_LDS
        __builtin_amdgcn_global_load_async_to_lds_b128(
            (gas_v4i_p)(wsrc0 + k0),
            (las_v4i_p)&lds_b[nbuf][b_row0][b_seg0], 0, 0);
        __builtin_amdgcn_global_load_async_to_lds_b128(
            (gas_v4i_p)(wsrc1 + k0),
            (las_v4i_p)&lds_b[nbuf][b_row0 + 64][b_seg0], 0, 0);
#else
        const uint4* wp = (const uint4*)(wrow + k0);
        b_reg[0] = wp[0];
        b_reg[1] = wp[1];
#endif
    };

    auto store_stage = [&](int buf) {
        bf16_t* pa = &lds_a[buf][ldrow][ldseg];
#pragma unroll
        for (int i = 0; i < 4; ++i)
            *(v4bf*)(pa + i * 4) = __builtin_convertvector(a_reg[i], v4bf);
#if !HAS_ASYNC_LDS
        uint4* pb = (uint4*)&lds_b[buf][ldrow][ldseg];
        pb[0] = b_reg[0];
        pb[1] = b_reg[1];
#endif
    };

    auto compute = [&](int buf) {
#pragma unroll
        for (int t = 0; t < 2; ++t) {
            const int arow = wm * 32 + t * 16 + lrow;
            // A frag: lanes 0-15 K=0..15, lanes 16-31 K=16..31
            v16bf a = mk16(*(const v8bf*)&lds_a[buf][arow][lhi * 16],
                           *(const v8bf*)&lds_a[buf][arow][lhi * 16 + 8]);
#pragma unroll
            for (int u = 0; u < 4; ++u) {
                const int brow = wn * 64 + u * 16 + lrow;
                // B frag: lanes 0-15 K=0..15, lanes 16-31 K=16..31
                v16bf b = mk16(*(const v8bf*)&lds_b[buf][brow][lhi * 16],
                               *(const v8bf*)&lds_b[buf][brow][lhi * 16 + 8]);
                acc[t][u] = __builtin_amdgcn_wmma_f32_16x16x32_bf16(
                    false, a, false, b, (short)0, acc[t][u], false, false);
            }
        }
    };

    // Prologue: stage k-step 0 into buffer 0.
    load_stage(0, 0);
    store_stage(0);
#if HAS_ASYNC_LDS
    wait_async_all();      // producer drains ASYNCcnt before barrier
#endif
    __syncthreads();

    const int STEPS = IN_F / 32;  // 128
    for (int s = 0; s < STEPS; ++s) {
        const int buf = s & 1;
        if (s + 1 < STEPS) load_stage((s + 1) * 32, buf ^ 1);  // prefetch
        compute(buf);                                          // WMMA on buf
        if (s + 1 < STEPS) store_stage(buf ^ 1);               // A regs -> lds
#if HAS_ASYNC_LDS
        wait_async_all();
#endif
        __syncthreads();
    }

    // Epilogue: C VGPR r holds M = r + 8*lhi; lanes give N = u*16 + lrow.
    float bl[4];
#pragma unroll
    for (int u = 0; u < 4; ++u)
        bl[u] = bias[n0 + wn * 64 + u * 16 + lrow];

#pragma unroll
    for (int t = 0; t < 2; ++t) {
#pragma unroll
        for (int u = 0; u < 4; ++u) {
            const int n = n0 + wn * 64 + u * 16 + lrow;
#pragma unroll
            for (int r = 0; r < 8; ++r) {
                const int m = m0 + wm * 32 + t * 16 + lhi * 8 + r;
                out[(size_t)m * OUT_F + n] = acc[t][u][r] + bl[u];
            }
        }
    }
}

// ---------------- host launcher ----------------

extern "C" void kernel_launch(void* const* d_in, const int* in_sizes, int n_in,
                              void* d_out, int out_size, void* d_ws, size_t ws_size,
                              hipStream_t stream) {
    const float* x    = (const float*)d_in[0];
    const float* wmu  = (const float*)d_in[1];
    const float* wrho = (const float*)d_in[2];
    const float* bmu  = (const float*)d_in[3];
    const float* brho = (const float*)d_in[4];
    const float* weps = (const float*)d_in[5];
    const float* beps = (const float*)d_in[6];
    float* out = (float*)d_out;

    // Workspace layout: [0, 32MB): W in bf16 ; [32MB, +16KB): bias f32
    bf16_t* wbf   = (bf16_t*)d_ws;
    float*  biasw = (float*)((char*)d_ws + (size_t)OUT_F * IN_F * sizeof(bf16_t));

    // 1) Sample weights -> bf16 (16.7M elems, 4 per thread)
    {
        size_t nthreads = (size_t)OUT_F * IN_F / 4;
        dim3 grid((unsigned)(nthreads / 256));
        prep_weight_kernel<<<grid, 256, 0, stream>>>(wmu, wrho, weps, wbf);
    }
    // 2) Sample bias -> f32
    prep_bias_kernel<<<dim3(OUT_F / 256), 256, 0, stream>>>(bmu, brho, beps, biasw);

    // 3) WMMA GEMM: out = x @ W^T + bias
    {
        dim3 grid(OUT_F / 128, BATCH / 128);  // (32, 64)
        gemm_wmma_kernel<<<grid, 256, 0, stream>>>(x, wbf, biasw, out);
    }
}
